// ViSNetBlock_52063593562438
// MI455X (gfx1250) — matrix operand order repaired and checked by reading, not compiled, not run on hardware
//
#include <hip/hip_runtime.h>
#include <hip/hip_bf16.h>
#include <math.h>

#define H       128
#define HEADS   8
#define HD      16
#define NRBF    32
#define CUT     5.0f
#define NLAYERS 2
#define PI_F    3.14159265358979f

typedef _Float16 hv16 __attribute__((ext_vector_type(16)));
typedef float    fv8  __attribute__((ext_vector_type(8)));

__device__ __forceinline__ float silu_f(float v) { return v / (1.0f + __expf(-v)); }

// ---------------------------------------------------------------------------
// WMMA GEMM:  C[M,N] = act( (A[M,K](f32->f16) @ B[K,N](f16 stored [N][K]) + bias) * rowscale )
// One wave computes a 16x64 tile (4 accumulators sharing one A fragment).
// 4 waves per block. K multiple of 32, M multiple of 16, N multiple of 64.
// ---------------------------------------------------------------------------
__global__ __launch_bounds__(128)
void k_wmma_gemm(const float* __restrict__ A, const _Float16* __restrict__ Bt,
                 const float* __restrict__ bias, const float* __restrict__ rowscale,
                 float* __restrict__ C, int M, int N, int K, int act)
{
    const int wave = threadIdx.x >> 5;
    const int lane = threadIdx.x & 31;
    const int nt_n = N >> 6;                       // 64-wide column tiles
    const int tile = blockIdx.x * 4 + wave;
    if (tile >= (M >> 4) * nt_n) return;
    const int mt = tile / nt_n;
    const int nt = tile - mt * nt_n;
    const int g  = lane >> 4;                      // lane half (K-split selector)
    const int li = lane & 15;
    const int arow = mt * 16 + li;                 // A row this lane feeds
    const int bcol = nt * 64 + li;                 // first B column this lane feeds

    const float*    ap  = A  + (size_t)arow * K + 16 * g;      // A layout: 16 contiguous K per half
    const _Float16* bp0 = Bt + (size_t)(bcol     ) * K + 16 * g; // B layout: 16 contiguous K per half, same as A
    const _Float16* bp1 = Bt + (size_t)(bcol + 16) * K + 16 * g;
    const _Float16* bp2 = Bt + (size_t)(bcol + 32) * K + 16 * g;
    const _Float16* bp3 = Bt + (size_t)(bcol + 48) * K + 16 * g;

    fv8 acc0 = {}, acc1 = {}, acc2 = {}, acc3 = {};
    for (int kk = 0; kk < K; kk += 32) {
        float4 a0 = *(const float4*)(ap + kk);
        float4 a1 = *(const float4*)(ap + kk + 4);
        float4 a2 = *(const float4*)(ap + kk + 8);
        float4 a3 = *(const float4*)(ap + kk + 12);
        hv16 a;
        a[0]=(_Float16)a0.x;  a[1]=(_Float16)a0.y;  a[2]=(_Float16)a0.z;  a[3]=(_Float16)a0.w;
        a[4]=(_Float16)a1.x;  a[5]=(_Float16)a1.y;  a[6]=(_Float16)a1.z;  a[7]=(_Float16)a1.w;
        a[8]=(_Float16)a2.x;  a[9]=(_Float16)a2.y;  a[10]=(_Float16)a2.z; a[11]=(_Float16)a2.w;
        a[12]=(_Float16)a3.x; a[13]=(_Float16)a3.y; a[14]=(_Float16)a3.z; a[15]=(_Float16)a3.w;
        hv16 b0 = *(const hv16*)(bp0 + kk);
        hv16 b1 = *(const hv16*)(bp1 + kk);
        hv16 b2 = *(const hv16*)(bp2 + kk);
        hv16 b3 = *(const hv16*)(bp3 + kk);
        acc0 = __builtin_amdgcn_wmma_f32_16x16x32_f16(false, a, false, b0, (short)0, acc0, false, false);
        acc1 = __builtin_amdgcn_wmma_f32_16x16x32_f16(false, a, false, b1, (short)0, acc1, false, false);
        acc2 = __builtin_amdgcn_wmma_f32_16x16x32_f16(false, a, false, b2, (short)0, acc2, false, false);
        acc3 = __builtin_amdgcn_wmma_f32_16x16x32_f16(false, a, false, b3, (short)0, acc3, false, false);
    }

    const int rbase = mt * 16 + 8 * g;
    float rs[8];
#pragma unroll
    for (int i = 0; i < 8; ++i) rs[i] = rowscale ? rowscale[rbase + i] : 1.0f;

    fv8 accs[4] = {acc0, acc1, acc2, acc3};
#pragma unroll
    for (int c = 0; c < 4; ++c) {
        const int col = bcol + 16 * c;
        const float bv = bias ? bias[col] : 0.0f;
#pragma unroll
        for (int i = 0; i < 8; ++i) {
            float v = accs[c][i] + bv;
            v *= rs[i];
            if (act) v = silu_f(v);
            C[(size_t)(rbase + i) * N + col] = v;
        }
    }
}

// ---------------------------------------------------------------------------
// Elementwise / gather-scatter kernels (fp32)
// ---------------------------------------------------------------------------
__global__ void k_transpose_f16(const float* __restrict__ S, _Float16* __restrict__ D, int K, int N) {
    int t = blockIdx.x * blockDim.x + threadIdx.x;
    if (t >= K * N) return;
    int k = t / N, n = t - k * N;
    D[(size_t)n * K + k] = (_Float16)S[(size_t)k * N + n];
}

__global__ void k_zero(float* __restrict__ p, int n) {
    int t = blockIdx.x * blockDim.x + threadIdx.x;
    if (t < n) p[t] = 0.0f;
}

__global__ void k_edge_geom(const float* __restrict__ pos, const int* __restrict__ ei,
                            const float* __restrict__ means, const float* __restrict__ betas,
                            float* __restrict__ dij, float* __restrict__ cr,
                            float* __restrict__ rbf, int E) {
    int e = blockIdx.x * blockDim.x + threadIdx.x;
    if (e >= E) return;
    int s = ei[e], d = ei[E + e];
    float ex = pos[s*3+0] - pos[d*3+0];
    float ey = pos[s*3+1] - pos[d*3+1];
    float ez = pos[s*3+2] - pos[d*3+2];
    float r  = sqrtf(ex*ex + ey*ey + ez*ez);
    float inv = 1.0f / r;
    dij[e*3+0] = ex*inv; dij[e*3+1] = ey*inv; dij[e*3+2] = ez*inv;
    float c = (r < CUT) ? 0.5f * (cosf(r * PI_F / CUT) + 1.0f) : 0.0f;
    cr[e] = c;
    float er = __expf(-r);               // alpha = 5/CUTOFF = 1
    for (int j = 0; j < NRBF; ++j) {
        float t = er - means[j];
        rbf[(size_t)e * NRBF + j] = c * __expf(-betas[j] * t * t);
    }
}

// gather emb[z] into first half of xcat (stride 2H)
__global__ void k_gather_emb(const float* __restrict__ emb, const int* __restrict__ z,
                             float* __restrict__ xcat, int n) {
    int t = blockIdx.x * blockDim.x + threadIdx.x;
    if (t >= n) return;
    int i = t / H, h = t - i * H;
    xcat[(size_t)i * 2 * H + h] = emb[(size_t)z[i] * H + h];
}

// agg = segment_sum(ne_emb[z[src]] * W, dst), accumulated into second half of xcat
__global__ void k_scatter_agg(const float* __restrict__ Wm, const float* __restrict__ ne_emb,
                              const int* __restrict__ z, const int* __restrict__ ei,
                              float* __restrict__ xcat, int E) {
    int t = blockIdx.x * blockDim.x + threadIdx.x;
    if (t >= E * H) return;
    int e = t / H, h = t - e * H;
    int s = ei[e], d = ei[E + e];
    atomicAdd(&xcat[(size_t)d * 2 * H + H + h], ne_emb[(size_t)z[s] * H + h] * Wm[t]);
}

__global__ void k_fij(const float* __restrict__ x, const float* __restrict__ ee,
                      const int* __restrict__ ei, float* __restrict__ fij, int E) {
    int t = blockIdx.x * blockDim.x + threadIdx.x;
    if (t >= E * H) return;
    int e = t / H, h = t - e * H;
    int s = ei[e], d = ei[E + e];
    fij[t] = (x[(size_t)d * H + h] + x[(size_t)s * H + h]) * ee[t];
}

__global__ void k_layernorm(const float* __restrict__ X, const float* __restrict__ g,
                            const float* __restrict__ b, float* __restrict__ Y, int n) {
    int i = blockIdx.x * blockDim.x + threadIdx.x;
    if (i >= n) return;
    const float* x = X + (size_t)i * H;
    float mu = 0.0f;
    for (int h = 0; h < H; ++h) mu += x[h];
    mu *= (1.0f / H);
    float var = 0.0f;
    for (int h = 0; h < H; ++h) { float d = x[h] - mu; var += d * d; }
    var *= (1.0f / H);
    float inv = rsqrtf(var + 1e-5f);
    float* y = Y + (size_t)i * H;
    for (int h = 0; h < H; ++h) y[h] = (x[h] - mu) * inv * g[h] + b[h];
}

// Y[t] = X[t] * w[t % H]
__global__ void k_mulrow(const float* __restrict__ X, const float* __restrict__ w,
                         float* __restrict__ Y, int n) {
    int t = blockIdx.x * blockDim.x + threadIdx.x;
    if (t >= n) return;
    Y[t] = X[t] * w[t % H];
}

__global__ void k_vecdot(const float* __restrict__ vproj, float* __restrict__ vdot, int n) {
    int t = blockIdx.x * blockDim.x + threadIdx.x;
    if (t >= n) return;
    int i = t / H, h = t - i * H;
    float acc = 0.0f;
    for (int sp = 0; sp < 3; ++sp) {
        const float* row = vproj + ((size_t)i * 3 + sp) * 384;
        acc += row[h] * row[H + h];
    }
    vdot[t] = acc;
}

__global__ void k_attn_vj(const float* __restrict__ q, const float* __restrict__ k,
                          const float* __restrict__ v, const float* __restrict__ dk,
                          const float* __restrict__ dv, const float* __restrict__ cr,
                          const int* __restrict__ ei, float* __restrict__ vj, int E) {
    int t = blockIdx.x * blockDim.x + threadIdx.x;
    if (t >= E * HEADS) return;
    int e = t / HEADS, hd = t - e * HEADS;
    int s = ei[e], d = ei[E + e];
    int c0 = hd * HD;
    float acc = 0.0f;
    for (int i = 0; i < HD; ++i) {
        int c = c0 + i;
        acc += q[(size_t)d * H + c] * k[(size_t)s * H + c] * dk[(size_t)e * H + c];
    }
    float a = silu_f(acc) * cr[e];
    for (int i = 0; i < HD; ++i) {
        int c = c0 + i;
        vj[(size_t)e * H + c] = v[(size_t)s * H + c] * dv[(size_t)e * H + c] * a;
    }
}

__global__ void k_scatter2(const float* __restrict__ vj, const float* __restrict__ sout,
                           const float* __restrict__ vl, const float* __restrict__ dij,
                           const int* __restrict__ ei, float* __restrict__ xagg,
                           float* __restrict__ vecagg, int E) {
    int t = blockIdx.x * blockDim.x + threadIdx.x;
    if (t >= E * H) return;
    int e = t / H, h = t - e * H;
    int s = ei[e], d = ei[E + e];
    atomicAdd(&xagg[(size_t)d * H + h], vj[t]);
    float s1 = sout[(size_t)e * 2 * H + h];
    float s2 = sout[(size_t)e * 2 * H + H + h];
    for (int sp = 0; sp < 3; ++sp) {
        float m = vl[((size_t)s * 3 + sp) * H + h] * s1 + s2 * dij[e * 3 + sp];
        atomicAdd(&vecagg[((size_t)d * 3 + sp) * H + h], m);
    }
}

// f_ij += f_out * [ sum_s wt*ws - (wt.d)(ws.d) ]   (w1.w2 with d unit vector)
__global__ void k_fupdate(const float* __restrict__ wt, const float* __restrict__ wsrc,
                          const float* __restrict__ fout, const float* __restrict__ dij,
                          const int* __restrict__ ei, float* __restrict__ fij, int E) {
    int t = blockIdx.x * blockDim.x + threadIdx.x;
    if (t >= E * H) return;
    int e = t / H, h = t - e * H;
    int s = ei[e], d = ei[E + e];
    float at = 0.0f, as = 0.0f, dot = 0.0f;
    for (int sp = 0; sp < 3; ++sp) {
        float a  = wt[((size_t)d * 3 + sp) * H + h];
        float b  = wsrc[((size_t)s * 3 + sp) * H + h];
        float dd = dij[e * 3 + sp];
        at += a * dd; as += b * dd; dot += a * b;
    }
    fij[t] += fout[t] * (dot - at * as);
}

__global__ void k_node_update(const float* __restrict__ oout, const float* __restrict__ vdot,
                              const float* __restrict__ vproj, const float* __restrict__ vagg,
                              float* __restrict__ x, float* __restrict__ vec, int n) {
    int t = blockIdx.x * blockDim.x + threadIdx.x;
    if (t >= n) return;
    int i = t / H, h = t - i * H;
    float o1 = oout[(size_t)i * 384 + h];
    float o2 = oout[(size_t)i * 384 + 128 + h];
    float o3 = oout[(size_t)i * 384 + 256 + h];
    x[t] += vdot[t] * o2 + o3;
    for (int sp = 0; sp < 3; ++sp) {
        size_t vi = ((size_t)i * 3 + sp) * H + h;
        vec[vi] += vproj[((size_t)i * 3 + sp) * 384 + 256 + h] * o1 + vagg[vi];
    }
}

__global__ void k_vec_out(const float* __restrict__ vec, const float* __restrict__ w,
                          float* __restrict__ out, int n) {
    int t = blockIdx.x * blockDim.x + threadIdx.x;
    if (t >= n) return;
    out[t] = vec[t] * w[t % H];
}

// ---------------------------------------------------------------------------
// Host orchestration
// ---------------------------------------------------------------------------
extern "C" void kernel_launch(void* const* d_in, const int* in_sizes, int n_in,
                              void* d_out, int out_size, void* d_ws, size_t ws_size,
                              hipStream_t stream) {
    (void)n_in; (void)out_size; (void)ws_size;
    const int*   z    = (const int*)d_in[0];
    const float* pos  = (const float*)d_in[1];
    const int*   ei   = (const int*)d_in[2];
    const int Nn = in_sizes[0];
    const int E  = in_sizes[2] / 2;
    auto P = [&](int i) { return (const float*)d_in[i]; };
    // param indices (dict order): 3 emb, 4 ne_emb, 5 ne_dist_w, 6 ne_dist_b, 7 ne_comb_w,
    // 8 ne_comb_b, 9 rbf_means, 10 rbf_betas, 11 ee_w, 12 ee_b, 13 ln_g, 14 ln_b, 15 vln_w,
    // 16 vec_w, 17 q_w, 18 q_b, 19 k_w, 20 k_b, 21 v_w, 22 v_b, 23 dk_w, 24 dk_b, 25 dv_w,
    // 26 dv_b, 27 s_w, 28 s_b, 29 o_w, 30 o_b, 31 f_w, 32 f_b, 33 wsrc_w, 34 wtrg_w,
    // 35 out_ln_g, 36 out_ln_b, 37 out_vln_w

    // ---- workspace carve-out (256B aligned) ----
    char* wsb = (char*)d_ws;
    size_t off = 0;
    auto allocF = [&](size_t elems) { off = (off + 255) & ~(size_t)255; float* p = (float*)(wsb + off); off += elems * 4; return p; };
    auto allocH = [&](size_t elems) { off = (off + 255) & ~(size_t)255; _Float16* p = (_Float16*)(wsb + off); off += elems * 2; return p; };

    _Float16* ne_dist_wt = allocH((size_t)128 * 32);
    _Float16* ne_comb_wt = allocH((size_t)128 * 256);
    _Float16* ee_wt      = allocH((size_t)128 * 32);
    _Float16 *vec_wt[NLAYERS], *q_wt[NLAYERS], *k_wt[NLAYERS], *v_wt[NLAYERS],
             *dk_wt[NLAYERS], *dv_wt[NLAYERS], *s_wt[NLAYERS], *o_wt[NLAYERS],
             *f_wt[NLAYERS], *wsrc_wt[NLAYERS], *wtrg_wt[NLAYERS];
    for (int l = 0; l < NLAYERS; ++l) {
        vec_wt[l] = allocH((size_t)384 * 128);
        q_wt[l]   = allocH((size_t)128 * 128);
        k_wt[l]   = allocH((size_t)128 * 128);
        v_wt[l]   = allocH((size_t)128 * 128);
        dk_wt[l]  = allocH((size_t)128 * 128);
        dv_wt[l]  = allocH((size_t)128 * 128);
        s_wt[l]   = allocH((size_t)256 * 128);
        o_wt[l]   = allocH((size_t)384 * 128);
        f_wt[l]   = allocH((size_t)128 * 128);
        wsrc_wt[l]= allocH((size_t)128 * 128);
        wtrg_wt[l]= allocH((size_t)128 * 128);
    }
    float* dijB  = allocF((size_t)E * 3);
    float* crB   = allocF((size_t)E);
    float* rbfB  = allocF((size_t)E * NRBF);
    float* xcatB = allocF((size_t)Nn * 2 * H);
    float* xB    = allocF((size_t)Nn * H);
    float* vecB  = allocF((size_t)Nn * 3 * H);
    float* fijB  = allocF((size_t)E * H);
    float* xlB   = allocF((size_t)Nn * H);
    float* vlB   = allocF((size_t)Nn * 3 * H);
    float* qB    = allocF((size_t)Nn * H);
    float* kB    = allocF((size_t)Nn * H);
    float* vB    = allocF((size_t)Nn * H);
    float* vprojB= allocF((size_t)Nn * 3 * 384);
    float* vdotB = allocF((size_t)Nn * H);
    float* eA    = allocF((size_t)E * H);   // W / dk / f_out
    float* eB    = allocF((size_t)E * H);   // ee / dv
    float* vjB   = allocF((size_t)E * H);
    float* soutB = allocF((size_t)E * 2 * H);
    float* xaggB = allocF((size_t)Nn * H);
    float* vaggB = allocF((size_t)Nn * 3 * H);
    float* ooutB = allocF((size_t)Nn * 384);
    float* wtB   = allocF((size_t)Nn * 3 * H);
    float* wsB   = allocF((size_t)Nn * 3 * H);

    auto g1 = [](int n) { return dim3((unsigned)((n + 255) / 256)); };
    auto T  = [&](const float* src, _Float16* dst, int K, int N) {
        k_transpose_f16<<<g1(K * N), 256, 0, stream>>>(src, dst, K, N);
    };
    auto GEMM = [&](const float* A, const _Float16* Bt, const float* bias, const float* rs,
                    float* C, int M, int N, int K, int act) {
        int tiles = (M / 16) * (N / 64);
        k_wmma_gemm<<<dim3((unsigned)((tiles + 3) / 4)), 128, 0, stream>>>(A, Bt, bias, rs, C, M, N, K, act);
    };

    // ---- weight prep (f32 [K][N] -> f16 [N][K]) ----
    T(P(5), ne_dist_wt, 32, 128);
    T(P(7), ne_comb_wt, 256, 128);
    T(P(11), ee_wt, 32, 128);
    for (int l = 0; l < NLAYERS; ++l) {
        T(P(16) + (size_t)l * 128 * 384, vec_wt[l], 128, 384);
        T(P(17) + (size_t)l * 128 * 128, q_wt[l],  128, 128);
        T(P(19) + (size_t)l * 128 * 128, k_wt[l],  128, 128);
        T(P(21) + (size_t)l * 128 * 128, v_wt[l],  128, 128);
        T(P(23) + (size_t)l * 128 * 128, dk_wt[l], 128, 128);
        T(P(25) + (size_t)l * 128 * 128, dv_wt[l], 128, 128);
        T(P(27) + (size_t)l * 128 * 256, s_wt[l],  128, 256);
        T(P(29) + (size_t)l * 128 * 384, o_wt[l],  128, 384);
        T(P(31) + (size_t)l * 128 * 128, f_wt[l],  128, 128);
        T(P(33) + (size_t)l * 128 * 128, wsrc_wt[l], 128, 128);
        T(P(34) + (size_t)l * 128 * 128, wtrg_wt[l], 128, 128);
    }

    // ---- edge geometry + RBF ----
    k_edge_geom<<<g1(E), 256, 0, stream>>>(pos, ei, P(9), P(10), dijB, crB, rbfB, E);

    // ---- neighbor embedding ----
    k_zero<<<g1(Nn * 2 * H), 256, 0, stream>>>(xcatB, Nn * 2 * H);
    k_gather_emb<<<g1(Nn * H), 256, 0, stream>>>(P(3), z, xcatB, Nn * H);
    GEMM(rbfB, ne_dist_wt, P(6), crB, eA, E, 128, 32, 0);              // W = (rbf@Wd+b)*Cr
    k_scatter_agg<<<g1(E * H), 256, 0, stream>>>(eA, P(4), z, ei, xcatB, E);
    GEMM(xcatB, ne_comb_wt, P(8), nullptr, xB, Nn, 128, 256, 0);       // x = [x,agg]@Wc+b

    // ---- edge embedding f_ij ----
    GEMM(rbfB, ee_wt, P(12), nullptr, eB, E, 128, 32, 0);
    k_fij<<<g1(E * H), 256, 0, stream>>>(xB, eB, ei, fijB, E);

    k_zero<<<g1(Nn * 3 * H), 256, 0, stream>>>(vecB, Nn * 3 * H);

    // ---- attention layers ----
    for (int l = 0; l < NLAYERS; ++l) {
        const int last = (l == NLAYERS - 1);
        k_layernorm<<<g1(Nn), 256, 0, stream>>>(xB, P(13) + l * H, P(14) + l * H, xlB, Nn);
        k_mulrow<<<g1(Nn * 3 * H), 256, 0, stream>>>(vecB, P(15) + l * H, vlB, Nn * 3 * H);

        GEMM(xlB, q_wt[l], P(18) + l * H, nullptr, qB, Nn, 128, 128, 0);
        GEMM(xlB, k_wt[l], P(20) + l * H, nullptr, kB, Nn, 128, 128, 0);
        GEMM(xlB, v_wt[l], P(22) + l * H, nullptr, vB, Nn, 128, 128, 0);
        GEMM(fijB, dk_wt[l], P(24) + l * H, nullptr, eA, E, 128, 128, 1);   // dk = silu(...)
        GEMM(fijB, dv_wt[l], P(26) + l * H, nullptr, eB, E, 128, 128, 1);   // dv = silu(...)
        GEMM(vlB, vec_wt[l], nullptr, nullptr, vprojB, Nn * 3, 384, 128, 0);
        k_vecdot<<<g1(Nn * H), 256, 0, stream>>>(vprojB, vdotB, Nn * H);

        k_attn_vj<<<g1(E * HEADS), 256, 0, stream>>>(qB, kB, vB, eA, eB, crB, ei, vjB, E);
        GEMM(vjB, s_wt[l], P(28) + l * 256, nullptr, soutB, E, 256, 128, 1); // silu(vj@s_w+b)

        k_zero<<<g1(Nn * H), 256, 0, stream>>>(xaggB, Nn * H);
        k_zero<<<g1(Nn * 3 * H), 256, 0, stream>>>(vaggB, Nn * 3 * H);
        k_scatter2<<<g1(E * H), 256, 0, stream>>>(vjB, soutB, vlB, dijB, ei, xaggB, vaggB, E);

        GEMM(xaggB, o_wt[l], P(30) + l * 384, nullptr, ooutB, Nn, 384, 128, 0);

        if (!last) {
            GEMM(vlB, wtrg_wt[l], nullptr, nullptr, wtB, Nn * 3, 128, 128, 0);
            GEMM(vlB, wsrc_wt[l], nullptr, nullptr, wsB, Nn * 3, 128, 128, 0);
            GEMM(fijB, f_wt[l], P(32) + l * H, nullptr, eA, E, 128, 128, 1); // f_out = silu(...)
            k_fupdate<<<g1(E * H), 256, 0, stream>>>(wtB, wsB, eA, dijB, ei, fijB, E);
        }
        k_node_update<<<g1(Nn * H), 256, 0, stream>>>(ooutB, vdotB, vprojB, vaggB, xB, vecB, Nn * H);
    }

    // ---- outputs: x (LN) then vec (scaled) ----
    float* out = (float*)d_out;
    k_layernorm<<<g1(Nn), 256, 0, stream>>>(xB, P(35), P(36), out, Nn);
    k_vec_out<<<g1(Nn * 3 * H), 256, 0, stream>>>(vecB, P(37), out + (size_t)Nn * H, Nn * 3 * H);
}
